// WindowAttention_63471026700562
// MI455X (gfx1250) — matrix operand order repaired and checked by reading, not compiled, not run on hardware
//
#include <hip/hip_runtime.h>
#include <hip/hip_bf16.h>

typedef __attribute__((ext_vector_type(16))) _Float16 v16h;
typedef __attribute__((ext_vector_type(8)))  _Float16 v8h;
typedef __attribute__((ext_vector_type(4)))  _Float16 v4h;
typedef __attribute__((ext_vector_type(8)))  float    v8f;
typedef __attribute__((ext_vector_type(4)))  float    v4f;

union ABFrag { v16h v; v8h h[2]; };

// Sizes
#define BWIN 4096
#define NTOK 49
#define CDIM 128
#define HEADS 4
#define DHEAD 32
#define NPAD 64
#define SCALE 0.17677669529663687f   // 1/sqrt(32)

// Workspace layout (halves / floats)
#define WQ_HALVES  98304   // 24 colTiles * 4 kTiles * 1024
#define WP_HALVES  32768   // 8 colTiles * 4 kTiles * 1024
#define BIASF_ELMS 16384   // 4 heads * 16 tiles * 32 lanes * 8

// LDS regions (in halves), total 32768 halves = 64 KB
#define X_OFF   0        // [64][128] f16 x            (recycled as P lo)
#define Q_OFF   8192     // [4][64][32] f16 q*scale    (recycled as P hi)
#define K_OFF   16384    // [4][64][32] f16 k          (recycled as O)
#define VT_OFF  24576    // [4][32][64] f16 v^T
#define P_OFF   0        // [4][64][64] f16 softmax(P)
#define O_OFF   16384    // [64][128]  f16 attn output

__device__ __forceinline__ v8f wmma_f16(const v16h& a, const v16h& b, const v8f& c) {
    return __builtin_amdgcn_wmma_f32_16x16x32_f16(false, a, false, b, (short)0, c, false, false);
}

// ---------------------------------------------------------------------------
// Prep: pack weights into WMMA B-fragment order (f16) + bias in C-fragment
// order (so the attention kernel can use it directly as the WMMA C operand).
// B-fragment tile (16x16 out, k=32): 1024 halves lane-major; lane l holds
// col n = tile_n*16 + (l&15), k = tile_k*32 + (l<16?0:16) + kk, kk=0..15.
// C-fragment tile: 256 floats as [lane][i]; lane l, vgpr i maps to
// row = m*16 + i + (l>>4)*8, col = c*16 + (l&15).
// ---------------------------------------------------------------------------
__global__ void prep_kernel(const float* __restrict__ qkv_w,
                            const float* __restrict__ proj_w,
                            const float* __restrict__ bias_table,
                            const int*   __restrict__ rel_index,
                            _Float16* __restrict__ wq,
                            _Float16* __restrict__ wp,
                            float*    __restrict__ biasf) {
    int gid = blockIdx.x * blockDim.x + threadIdx.x;
    if (gid < WQ_HALVES) {
        int bi = gid >> 10, r = gid & 1023;
        int lane = r >> 4, kk = r & 15;
        int c = bi >> 2, t = bi & 3;
        int k = t * 32 + ((lane < 16) ? 0 : 16) + kk;
        int n = c * 16 + (lane & 15);
        wq[gid] = (_Float16)qkv_w[k * 384 + n];
    } else if (gid < WQ_HALVES + WP_HALVES) {
        int idx = gid - WQ_HALVES;
        int bi = idx >> 10, r = idx & 1023;
        int lane = r >> 4, kk = r & 15;
        int c = bi >> 2, t = bi & 3;
        int k = t * 32 + ((lane < 16) ? 0 : 16) + kk;
        int n = c * 16 + (lane & 15);
        wp[idx] = (_Float16)proj_w[k * 128 + n];
    } else if (gid < WQ_HALVES + WP_HALVES + BIASF_ELMS) {
        int idx  = gid - (WQ_HALVES + WP_HALVES);
        int h    = idx >> 12;
        int rem  = idx & 4095;
        int tile = rem >> 8;           // m*4 + c
        int lane = (rem >> 3) & 31;
        int i    = rem & 7;
        int row  = (tile >> 2) * 16 + i + (lane >> 4) * 8;
        int col  = (tile & 3) * 16 + (lane & 15);
        float v  = -1e30f;             // mask padded key columns
        if (row < NTOK && col < NTOK)
            v = bias_table[rel_index[row * NTOK + col] * HEADS + h];
        biasf[idx] = v;
    }
}

// ---------------------------------------------------------------------------
// Fused window attention: one workgroup (8 wave32) per window.
// ---------------------------------------------------------------------------
__global__ __launch_bounds__(256)
void winattn_kernel(const float* __restrict__ x,
                    const float* __restrict__ qkv_b,
                    const float* __restrict__ proj_b,
                    const _Float16* __restrict__ wq,
                    const _Float16* __restrict__ wp,
                    const float* __restrict__ biasf,
                    float* __restrict__ out) {
    __shared__ _Float16 sm[32768];   // 64 KB, regions recycled (see defines)

    const int b    = blockIdx.x;
    const int tid  = threadIdx.x;
    // Scalarized wave index -> SALU branches instead of exec-mask predication
    const int wave = __builtin_amdgcn_readfirstlane(tid >> 5);
    const int lane = tid & 31;
    const int hb   = lane >> 4;      // which 16-lane half of the wave
    const int lr   = lane & 15;

    // ---- Phase 0: x (fp32, 49x128) -> LDS f16 [64][128], zero-padded rows.
    // Vectorized: float4 global loads (b128) + packed f16 b64 LDS stores.
    const float* xb = x + (size_t)b * NTOK * CDIM;
#pragma unroll
    for (int j = 0; j < 8; ++j) {
        const int s   = tid + j * 256;       // float4 slot, 2048 total
        const int row = s >> 5;              // 32 float4 per row
        v4f xv = {};
        if (row < NTOK) xv = *(const v4f*)(xb + s * 4);
        v4h hv;
        hv[0] = (_Float16)xv[0];
        hv[1] = (_Float16)xv[1];
        hv[2] = (_Float16)xv[2];
        hv[3] = (_Float16)xv[3];
        *(v4h*)&sm[X_OFF + s * 4] = hv;
    }
    __syncthreads();

    // ---- Phase 1: QKV GEMM: [64x128] @ [128x384]; wave w -> colTiles 3w..3w+2
    for (int m = 0; m < 4; ++m) {
        // A fragments for this row tile, all 4 k-steps (reused across 3 colTiles)
        ABFrag afr[4];
#pragma unroll
        for (int t = 0; t < 4; ++t) {
            const _Float16* xrow = &sm[X_OFF + (m * 16 + lr) * CDIM + t * 32 + hb * 16];
            afr[t].h[0] = *(const v8h*)(xrow);
            afr[t].h[1] = *(const v8h*)(xrow + 8);
        }
        for (int c3 = 0; c3 < 3; ++c3) {
            const int c = wave * 3 + c3;               // scalar
            v8f acc = {};
#pragma unroll
            for (int t = 0; t < 4; ++t) {
                v16h bf = *(const v16h*)(wq + (c * 4 + t) * 1024 + lane * 16);
                acc = wmma_f16(afr[t].v, bf, acc);
            }
            const float qb = qkv_b[c * 16 + lr];
            const int which = c >> 3;                  // 0=q 1=k 2=v (scalar)
            const int rem   = c & 7;
            const int h     = rem >> 1;
            const int dh    = (rem & 1) * 16;
            if (which == 0) {
#pragma unroll
                for (int i = 0; i < 8; ++i) {
                    const int row = m * 16 + i + hb * 8;
                    sm[Q_OFF + h * 2048 + row * 32 + dh + lr] =
                        (_Float16)((acc[i] + qb) * SCALE);
                }
            } else if (which == 1) {
#pragma unroll
                for (int i = 0; i < 8; ++i) {
                    const int row = m * 16 + i + hb * 8;
                    sm[K_OFF + h * 2048 + row * 32 + dh + lr] = (_Float16)(acc[i] + qb);
                }
            } else {
#pragma unroll
                for (int i = 0; i < 8; ++i) {
                    const int row = m * 16 + i + hb * 8;
                    sm[VT_OFF + h * 2048 + (dh + lr) * 64 + row] = (_Float16)(acc[i] + qb);
                }
            }
        }
    }
    __syncthreads();

    // ---- Phase 2: S = Q K^T + bias (bias preloaded as WMMA C operand),
    //      then softmax (wave h handles head h)
    v8f accS[4][4];
    if (wave < HEADS) {
        const int h = wave;
        for (int m = 0; m < 4; ++m) {
            ABFrag a;
            const _Float16* qrow = &sm[Q_OFF + h * 2048 + (m * 16 + lr) * 32 + hb * 16];
            a.h[0] = *(const v8h*)(qrow);
            a.h[1] = *(const v8h*)(qrow + 8);
#pragma unroll
            for (int c = 0; c < 4; ++c) {
                // C operand = relative-position bias, fragment-packed by prep
                v8f cb = *(const v8f*)(biasf + h * 4096 + (m * 4 + c) * 256 + lane * 8);
                v16h bf = *(const v16h*)&sm[K_OFF + h * 2048 + (c * 16 + lr) * 32 + hb * 16];
                accS[m][c] = wmma_f16(a.v, bf, cb);
            }
        }
        // row softmax: 4 in-lane cols, then 16-lane half reduction
        for (int m = 0; m < 4; ++m)
#pragma unroll
            for (int i = 0; i < 8; ++i) {
                float mx = accS[m][0][i];
#pragma unroll
                for (int c = 1; c < 4; ++c) mx = fmaxf(mx, accS[m][c][i]);
#pragma unroll
                for (int off = 1; off < 16; off <<= 1)
                    mx = fmaxf(mx, __shfl_xor(mx, off, 32));
                float s = 0.0f;
#pragma unroll
                for (int c = 0; c < 4; ++c) {
                    float e = __expf(accS[m][c][i] - mx);
                    accS[m][c][i] = e;
                    s += e;
                }
#pragma unroll
                for (int off = 1; off < 16; off <<= 1)
                    s += __shfl_xor(s, off, 32);
                const float inv = 1.0f / s;
#pragma unroll
                for (int c = 0; c < 4; ++c) accS[m][c][i] *= inv;
            }
    }
    __syncthreads();   // Q/K reads complete before P overwrites X∪Q region

    if (wave < HEADS) {
        const int h = wave;
        for (int m = 0; m < 4; ++m)
#pragma unroll
            for (int c = 0; c < 4; ++c)
#pragma unroll
                for (int i = 0; i < 8; ++i) {
                    const int row = m * 16 + i + hb * 8;
                    sm[P_OFF + h * 4096 + row * 64 + c * 16 + lr] = (_Float16)accS[m][c][i];
                }
    }
    __syncthreads();

    // ---- Phase 3: O_h = P_h @ V_h  (K region recycled as O [64][128] f16)
    if (wave < HEADS) {
        const int h = wave;
        for (int m = 0; m < 4; ++m) {
            ABFrag a[2];
#pragma unroll
            for (int t = 0; t < 2; ++t) {
                const _Float16* prow = &sm[P_OFF + h * 4096 + (m * 16 + lr) * 64 + t * 32 + hb * 16];
                a[t].h[0] = *(const v8h*)(prow);
                a[t].h[1] = *(const v8h*)(prow + 8);
            }
#pragma unroll
            for (int dt = 0; dt < 2; ++dt) {
                v8f acc = {};
#pragma unroll
                for (int t = 0; t < 2; ++t) {
                    v16h bf = *(const v16h*)&sm[VT_OFF + h * 2048 + (dt * 16 + lr) * 64
                                                + t * 32 + hb * 16];
                    acc = wmma_f16(a[t].v, bf, acc);
                }
#pragma unroll
                for (int i = 0; i < 8; ++i) {
                    const int row = m * 16 + i + hb * 8;
                    sm[O_OFF + row * CDIM + h * 32 + dt * 16 + lr] = (_Float16)acc[i];
                }
            }
        }
    }
    __syncthreads();

    // ---- Phase 4: out = O @ proj_w + proj_b; wave w -> colTile w (scalar)
    {
        const int c = wave;
        float* ob = out + (size_t)b * NTOK * CDIM;
        for (int m = 0; m < 4; ++m) {
            ABFrag afr[4];
#pragma unroll
            for (int t = 0; t < 4; ++t) {
                const _Float16* orow = &sm[O_OFF + (m * 16 + lr) * CDIM + t * 32 + hb * 16];
                afr[t].h[0] = *(const v8h*)(orow);
                afr[t].h[1] = *(const v8h*)(orow + 8);
            }
            v8f acc = {};
#pragma unroll
            for (int t = 0; t < 4; ++t) {
                v16h bf = *(const v16h*)(wp + (c * 4 + t) * 1024 + lane * 16);
                acc = wmma_f16(afr[t].v, bf, acc);
            }
            const float pb = proj_b[c * 16 + lr];
#pragma unroll
            for (int i = 0; i < 8; ++i) {
                const int row = m * 16 + i + hb * 8;
                if (row < NTOK)
                    ob[row * CDIM + c * 16 + lr] = acc[i] + pb;
            }
        }
    }
}

extern "C" void kernel_launch(void* const* d_in, const int* in_sizes, int n_in,
                              void* d_out, int out_size, void* d_ws, size_t ws_size,
                              hipStream_t stream) {
    (void)in_sizes; (void)n_in; (void)out_size; (void)ws_size;
    const float* x          = (const float*)d_in[0];
    const float* qkv_w      = (const float*)d_in[1];
    const float* qkv_b      = (const float*)d_in[2];
    const float* proj_w     = (const float*)d_in[3];
    const float* proj_b     = (const float*)d_in[4];
    const float* bias_table = (const float*)d_in[5];
    const int*   rel_index  = (const int*)d_in[6];

    _Float16* wq    = (_Float16*)d_ws;
    _Float16* wp    = wq + WQ_HALVES;
    float*    biasf = (float*)(wp + WP_HALVES);

    const int prep_total = WQ_HALVES + WP_HALVES + BIASF_ELMS;
    prep_kernel<<<(prep_total + 255) / 256, 256, 0, stream>>>(
        qkv_w, proj_w, bias_table, rel_index, wq, wp, biasf);

    winattn_kernel<<<BWIN, 256, 0, stream>>>(
        x, qkv_b, proj_b, wq, wp, biasf, (float*)d_out);
}
